// MultiHeadAttention_59837484368177
// MI455X (gfx1250) — matrix operand order repaired and checked by reading, not compiled, not run on hardware
//
#include <hip/hip_runtime.h>
#include <hip/hip_bf16.h>
#include <cstdint>

typedef __bf16 bf16_t;
typedef __attribute__((ext_vector_type(16))) __bf16 v16bf;
typedef __attribute__((ext_vector_type(8)))  __bf16 v8bf;
typedef __attribute__((ext_vector_type(8)))  float  v8f;

#define B_  4
#define S_  2048
#define D_  1024
#define H_  16
#define HD_ 64
#define LDT 72   // padded LDS tile pitch (144B rows: 16B aligned, bank-spread)

// ---------------------------------------------------------------------------
// Fragment loaders (CDNA5 wave32 WMMA layouts, 16-bit data)
// A 16x32 (MxK): lane&15 = row M; v16bf [0..7]=K(8*hi+0..7), [8..15]=K(16+8*hi+0..7)
// B 32x16 (KxN): lane&15 = col N; v16bf [0..7]=K(8*hi+0..7), [8..15]=K(16+8*hi+0..7)
// C/D 16x16 f32: VGPR r, lane l -> (M = r + 8*(l>>4), N = l&15)
// ---------------------------------------------------------------------------
__device__ __forceinline__ v16bf load_a16x32(const bf16_t* base, int hi) {
  v8bf x = *(const v8bf*)(base + 8 * hi);
  v8bf y = *(const v8bf*)(base + 16 + 8 * hi);
  v16bf r;
#pragma unroll
  for (int i = 0; i < 8; ++i) { r[i] = x[i]; r[i + 8] = y[i]; }
  return r;
}

__device__ __forceinline__ v16bf load_b32x16(const bf16_t* base, int hi) {
  v8bf x = *(const v8bf*)(base + 8 * hi);
  v8bf y = *(const v8bf*)(base + 16 + 8 * hi);
  v16bf r;
#pragma unroll
  for (int i = 0; i < 8; ++i) { r[i] = x[i]; r[i + 8] = y[i]; }
  return r;
}

__device__ __forceinline__ v8f wmma_bf16f32(v16bf a, v16bf b, v8f c) {
  return __builtin_amdgcn_wmma_f32_16x16x32_bf16(
      false, a, false, b, (short)0, c, false, false);
}

__device__ __forceinline__ float rowmax16(float v) {
#pragma unroll
  for (int m = 1; m <= 8; m <<= 1) v = fmaxf(v, __shfl_xor(v, m, 32));
  return v;
}
__device__ __forceinline__ float rowsum16(float v) {
#pragma unroll
  for (int m = 1; m <= 8; m <<= 1) v += __shfl_xor(v, m, 32);
  return v;
}

// ---------------------------------------------------------------------------
// Async 64x64 bf16 tile copy: global (row-major, gRowStride elems) -> LDS
// (pitch LDT). 128 threads x 4 x b128 chunks = 8KB; tracked on ASYNCcnt
// (4 async instructions per wave). Data bypasses VGPRs entirely.
// ---------------------------------------------------------------------------
__device__ __forceinline__ void async_stage_64x64(const bf16_t* g,
                                                  size_t gRowStride,
                                                  bf16_t* lds, int tid) {
#pragma unroll
  for (int k2 = 0; k2 < 4; ++k2) {
    const int c = tid + 128 * k2;       // chunk id 0..511
    const int row = c >> 3;             // 64 rows
    const int col = (c & 7) * 8;        // 8 x 16B chunks per row
    unsigned long long ga =
        (unsigned long long)(uintptr_t)(g + (size_t)row * gRowStride + col);
    unsigned la = (unsigned)(uintptr_t)(lds + row * LDT + col);
    asm volatile("global_load_async_to_lds_b128 %0, %1, off"
                 :: "v"(la), "v"(ga) : "memory");
  }
}

// ---------------------------------------------------------------------------
// fp32 -> bf16 convert (grid-stride)
// ---------------------------------------------------------------------------
__global__ void cvt_f32_to_bf16(const float* __restrict__ in,
                                bf16_t* __restrict__ out, int n) {
  int i = blockIdx.x * blockDim.x + threadIdx.x;
  int stride = gridDim.x * blockDim.x;
  for (; i < n; i += stride) out[i] = (bf16_t)in[i];
}

// ---------------------------------------------------------------------------
// Per-batch transpose: Vt[b*D + dcol][s] = Vp[b*S + s][dcol]  (bf16)
// 32x32 LDS tiles, 256 threads.
// ---------------------------------------------------------------------------
__global__ __launch_bounds__(256) void transpose_v(const bf16_t* __restrict__ in,
                                                   bf16_t* __restrict__ out) {
  __shared__ bf16_t tile[32][33];
  const int b = blockIdx.z;
  const int s0 = blockIdx.x * 32, d0 = blockIdx.y * 32;
  const int tx = threadIdx.x & 31, ty = threadIdx.x >> 5;  // 32 x 8
#pragma unroll
  for (int i = 0; i < 32; i += 8)
    tile[ty + i][tx] = in[((size_t)b * S_ + s0 + ty + i) * D_ + d0 + tx];
  __syncthreads();
#pragma unroll
  for (int i = 0; i < 32; i += 8)
    out[((size_t)b * D_ + d0 + ty + i) * S_ + s0 + tx] = tile[tx][ty + i];
}

// ---------------------------------------------------------------------------
// Y[M,N] = X[M,K] @ W[N,K]^T   (torch Linear convention), bf16 in, f32 acc
// 256 threads = 8 waves; block tile 128x128; wave tile 32x64.
// ---------------------------------------------------------------------------
template <bool OUT_BF16>
__global__ __launch_bounds__(256) void gemm_xwt(const bf16_t* __restrict__ X,
                                                const bf16_t* __restrict__ W,
                                                void* __restrict__ Yv,
                                                int M, int N, int K) {
  const int lane = threadIdx.x & 31;
  const int wave = threadIdx.x >> 5;
  const int hi   = lane >> 4;
  const int l15  = lane & 15;
  const int wm = wave >> 1;
  const int wn = wave & 1;
  const int rowBase = blockIdx.y * 128 + wm * 32;
  const int colBase = blockIdx.x * 128 + wn * 64;

  v8f c[2][4];
#pragma unroll
  for (int i = 0; i < 2; ++i)
#pragma unroll
    for (int j = 0; j < 4; ++j)
#pragma unroll
      for (int r = 0; r < 8; ++r) c[i][j][r] = 0.0f;

  const bf16_t* xr0 = X + (size_t)(rowBase + l15) * K;
  const bf16_t* xr1 = X + (size_t)(rowBase + 16 + l15) * K;
  const bf16_t* wr[4];
#pragma unroll
  for (int j = 0; j < 4; ++j) wr[j] = W + (size_t)(colBase + 16 * j + l15) * K;

  for (int k0 = 0; k0 < K; k0 += 32) {
    __builtin_prefetch(xr0 + k0 + 256, 0, 0);
    v16bf a0 = load_a16x32(xr0 + k0, hi);
    v16bf a1 = load_a16x32(xr1 + k0, hi);
#pragma unroll
    for (int j = 0; j < 4; ++j) {
      v16bf b = load_b32x16(wr[j] + k0, hi);
      c[0][j] = wmma_bf16f32(a0, b, c[0][j]);
      c[1][j] = wmma_bf16f32(a1, b, c[1][j]);
    }
  }

#pragma unroll
  for (int i = 0; i < 2; ++i)
#pragma unroll
    for (int j = 0; j < 4; ++j)
#pragma unroll
      for (int r = 0; r < 8; ++r) {
        const int row = rowBase + 16 * i + r + 8 * hi;
        const int col = colBase + 16 * j + l15;
        if constexpr (OUT_BF16)
          ((bf16_t*)Yv)[(size_t)row * N + col] = (bf16_t)c[i][j][r];
        else
          ((float*)Yv)[(size_t)row * N + col] = c[i][j][r];
      }
}

// ---------------------------------------------------------------------------
// Flash attention: block = (b, h, 64-query tile); 4 waves x 16 q-rows.
// K and V^T tiles double-buffered in LDS via async-to-LDS DMA; scores and
// P@V on v_wmma_f32_16x16x32_bf16; masked streaming softmax in registers.
// ---------------------------------------------------------------------------
__global__ __launch_bounds__(128) void attn_kernel(
    const bf16_t* __restrict__ Qp, const bf16_t* __restrict__ Kp,
    const bf16_t* __restrict__ Vtg, const int* __restrict__ mask,
    bf16_t* __restrict__ Ctx) {
  __shared__ bf16_t Ks[2][64 * LDT];   // key-tile: row = key j, col = d
  __shared__ bf16_t Vs[2][64 * LDT];   // V^T tile: row = d,    col = key j
  __shared__ bf16_t Ps[4][16 * LDT];   // per-wave P scratch (16 q x 64 keys)

  const int tid  = threadIdx.x;
  const int lane = tid & 31;
  const int wave = tid >> 5;
  const int hi   = lane >> 4;
  const int l15  = lane & 15;
  const int qBase = blockIdx.x * 64;
  const int h = blockIdx.y;
  const int b = blockIdx.z;
  const size_t bS = (size_t)b * S_;

  const bf16_t* Kbase = Kp + bS * D_ + h * HD_;                 // + j*D_
  const bf16_t* Vbase = Vtg + ((size_t)b * D_ + h * HD_) * S_;  // + d*S_ + j

  // Q A-fragments for this wave's 16-row strip (live across whole loop)
  const bf16_t* qptr = Qp + (bS + qBase + wave * 16 + l15) * D_ + h * HD_;
  v16bf aq0 = load_a16x32(qptr, hi);
  v16bf aq1 = load_a16x32(qptr + 32, hi);

  v8f O[4];
  float mrun[8], lrun[8];
#pragma unroll
  for (int t = 0; t < 4; ++t)
#pragma unroll
    for (int r = 0; r < 8; ++r) O[t][r] = 0.0f;
#pragma unroll
  for (int r = 0; r < 8; ++r) { mrun[r] = -1e30f; lrun[r] = 0.0f; }

  // prologue: asynchronously stage tile 0
  async_stage_64x64(Kbase, D_, &Ks[0][0], tid);
  async_stage_64x64(Vbase, S_, &Vs[0][0], tid);

  constexpr int NT = S_ / 64;  // 32 key tiles
  for (int it = 0; it < NT; ++it) {
    const int jBase = it * 64;
    const int cur = it & 1;

    if (it + 1 < NT) {
      // issue next tile into the other buffer, then wait for current tile
      const int nj = jBase + 64;
      async_stage_64x64(Kbase + (size_t)nj * D_, D_, &Ks[cur ^ 1][0], tid);
      async_stage_64x64(Vbase + nj, S_, &Vs[cur ^ 1][0], tid);
      asm volatile("s_wait_asynccnt 0x8" ::: "memory");  // tile `it` landed
    } else {
      asm volatile("s_wait_asynccnt 0x0" ::: "memory");
    }
    __syncthreads();

    const bf16_t* KsT = &Ks[cur][0];
    const bf16_t* VsT = &Vs[cur][0];

    // --- scores: S = Q @ K^T over hd=64 (2 K-steps), 4 key sub-tiles ---
    v8f sc[4];
#pragma unroll
    for (int t = 0; t < 4; ++t)
#pragma unroll
      for (int r = 0; r < 8; ++r) sc[t][r] = 0.0f;
#pragma unroll
    for (int ks = 0; ks < 2; ++ks) {
      v16bf pa = ks ? aq1 : aq0;
#pragma unroll
      for (int t = 0; t < 4; ++t) {
        v16bf bk = load_b32x16(&KsT[(t * 16 + l15) * LDT + ks * 32], hi);
        sc[t] = wmma_bf16f32(pa, bk, sc[t]);
      }
    }

    // --- scale + mask (mask==1 -> -1e9, matching reference) ---
#pragma unroll
    for (int t = 0; t < 4; ++t)
#pragma unroll
      for (int r = 0; r < 8; ++r) {
        const int qr = qBase + wave * 16 + r + 8 * hi;
        const int kc = jBase + t * 16 + l15;
        const int mv = mask[(bS + qr) * S_ + kc];
        const float v = sc[t][r] * 0.125f;  // 1/sqrt(64)
        sc[t][r] = (mv == 1) ? -1e9f : v;
      }

    // --- streaming softmax update ---
    float mnew[8], corr[8];
#pragma unroll
    for (int r = 0; r < 8; ++r) {
      float v = fmaxf(fmaxf(sc[0][r], sc[1][r]), fmaxf(sc[2][r], sc[3][r]));
      v = rowmax16(v);
      mnew[r] = fmaxf(mrun[r], v);
      corr[r] = __expf(mrun[r] - mnew[r]);
      mrun[r] = mnew[r];
    }
#pragma unroll
    for (int t = 0; t < 4; ++t)
#pragma unroll
      for (int r = 0; r < 8; ++r) sc[t][r] = __expf(sc[t][r] - mnew[r]);
#pragma unroll
    for (int r = 0; r < 8; ++r) {
      float s = (sc[0][r] + sc[1][r]) + (sc[2][r] + sc[3][r]);
      s = rowsum16(s);
      lrun[r] = lrun[r] * corr[r] + s;
    }
#pragma unroll
    for (int t = 0; t < 4; ++t)
#pragma unroll
      for (int r = 0; r < 8; ++r) O[t][r] *= corr[r];

    // --- re-layout P: D-fragment -> per-wave LDS -> A-fragment ---
    bf16_t* pbuf = &Ps[wave][0];
#pragma unroll
    for (int t = 0; t < 4; ++t)
#pragma unroll
      for (int r = 0; r < 8; ++r)
        pbuf[(r + 8 * hi) * LDT + t * 16 + l15] = (bf16_t)sc[t][r];
    asm volatile("s_wait_dscnt 0x0" ::: "memory");  // same-wave DS in-order
    v16bf pa0 = load_a16x32(pbuf + l15 * LDT, hi);
    v16bf pa1 = load_a16x32(pbuf + l15 * LDT + 32, hi);

    // --- O += P @ V (K-dim = 64 keys, 2 K-steps; 4 d sub-tiles) ---
#pragma unroll
    for (int ks = 0; ks < 2; ++ks) {
      v16bf pa = ks ? pa1 : pa0;
#pragma unroll
      for (int t = 0; t < 4; ++t) {
        v16bf bv = load_b32x16(&VsT[(t * 16 + l15) * LDT + ks * 32], hi);
        O[t] = wmma_bf16f32(pa, bv, O[t]);
      }
    }
    __syncthreads();  // all waves done with buffer `cur` before it is re-staged
  }

  // --- normalize and store context (bf16) ---
#pragma unroll
  for (int r = 0; r < 8; ++r) {
    const int qr = qBase + wave * 16 + r + 8 * hi;
    const float inv = 1.0f / lrun[r];
    bf16_t* crow = Ctx + (bS + qr) * D_ + h * HD_;
#pragma unroll
    for (int t = 0; t < 4; ++t) crow[t * 16 + l15] = (bf16_t)(O[t][r] * inv);
  }
}

// ---------------------------------------------------------------------------
// Host-side launch
// ---------------------------------------------------------------------------
extern "C" void kernel_launch(void* const* d_in, const int* in_sizes, int n_in,
                              void* d_out, int out_size, void* d_ws,
                              size_t ws_size, hipStream_t stream) {
  (void)in_sizes; (void)n_in; (void)out_size; (void)ws_size;
  const float* query = (const float*)d_in[0];
  const float* key_  = (const float*)d_in[1];
  const float* value = (const float*)d_in[2];
  const int*   mask  = (const int*)d_in[3];
  const float* wq = (const float*)d_in[4];
  const float* wk = (const float*)d_in[5];
  const float* wv = (const float*)d_in[6];
  const float* wo = (const float*)d_in[7];

  const int M = B_ * S_;              // 8192
  const size_t WB = (size_t)D_ * D_;  // 1M elems
  const size_t XB = (size_t)M * D_;   // 8M elems

  char* ws = (char*)d_ws;
  bf16_t* wqb = (bf16_t*)ws;
  bf16_t* wkb = wqb + WB;
  bf16_t* wvb = wkb + WB;
  bf16_t* wob = wvb + WB;
  bf16_t* qb  = wob + WB;
  bf16_t* kb  = qb + XB;
  bf16_t* vb  = kb + XB;
  bf16_t* Qp  = vb + XB;
  bf16_t* Kp  = Qp + XB;
  bf16_t* Vp  = Kp + XB;
  bf16_t* Vtg = Vp + XB;   // V^T per batch: [b*D + dcol][s]
  bf16_t* Ctx = qb;        // qb is dead after Qp is produced; reuse

  cvt_f32_to_bf16<<<4096, 256, 0, stream>>>(wq, wqb, (int)WB);
  cvt_f32_to_bf16<<<4096, 256, 0, stream>>>(wk, wkb, (int)WB);
  cvt_f32_to_bf16<<<4096, 256, 0, stream>>>(wv, wvb, (int)WB);
  cvt_f32_to_bf16<<<4096, 256, 0, stream>>>(wo, wob, (int)WB);
  cvt_f32_to_bf16<<<8192, 256, 0, stream>>>(query, qb, (int)XB);
  cvt_f32_to_bf16<<<8192, 256, 0, stream>>>(key_,  kb, (int)XB);
  cvt_f32_to_bf16<<<8192, 256, 0, stream>>>(value, vb, (int)XB);

  dim3 gg(D_ / 128, M / 128);  // (8, 64)
  gemm_xwt<true><<<gg, 256, 0, stream>>>(qb, wqb, Qp, M, D_, D_);
  gemm_xwt<true><<<gg, 256, 0, stream>>>(kb, wkb, Kp, M, D_, D_);
  gemm_xwt<true><<<gg, 256, 0, stream>>>(vb, wvb, Vp, M, D_, D_);

  dim3 gt(S_ / 32, D_ / 32, B_);  // (64, 32, 4)
  transpose_v<<<gt, 256, 0, stream>>>(Vp, Vtg);

  dim3 ga(S_ / 64, H_, B_);  // (32, 16, 4)
  attn_kernel<<<ga, 128, 0, stream>>>(Qp, Kp, Vtg, mask, Ctx);

  gemm_xwt<false><<<gg, 256, 0, stream>>>(Ctx, wob, d_out, M, D_, D_);
}
